// SpeakerBot_39711267619276
// MI455X (gfx1250) — matrix operand order repaired and checked by reading, not compiled, not run on hardware
//
#include <hip/hip_runtime.h>
#include <hip/hip_bf16.h>
#include <stdint.h>

// ---------------------------------------------------------------------------
// SpeakerBot RNN + Gumbel straight-through sampler for MI455X (gfx1250).
//
// Key insight: the scan carry is the STRAIGHT-THROUGH one-hot 'predict', so
//   h_in @ W_hh^T  ==  W_hh[:, argmax_prev]   (a gather, not a GEMM)
// and every batch row is independent across all 32 steps -> a persistent
// block walks its 64-row tile through the whole recurrence with zero global
// sync. The only dense GEMM is logits = h @ W_out^T (68.7 GFLOP total),
// done with v_wmma_f32_16x16x32_bf16 from LDS-resident W_out (128 KB bf16).
// HBM floor = 512 MB of one-hot 'message' stores (~23 us @ 23.3 TB/s).
// ---------------------------------------------------------------------------

#define NUM_STEPS 32
#define BATCH     16384
#define IN_DIM    18
#define HDIM      256
#define OUTDIM    256
#define TILE      64            // batch rows per block
#define WP        264           // LDS pitch (bf16 elems) for W_out  (16B-multiple rows)
#define HP        264           // LDS pitch (bf16 elems) for h tile
#define LP        257           // LDS pitch (f32 elems) for logits (bank-conflict-free column scans)

typedef __bf16 bf16_t;
typedef __attribute__((ext_vector_type(8)))  __bf16 v8bf;
typedef __attribute__((ext_vector_type(16))) __bf16 v16bf;
typedef __attribute__((ext_vector_type(8)))  float  v8f;
typedef __attribute__((ext_vector_type(4)))  float  v4f;

// counter-based hash -> Gumbel(0,1) noise (JAX threefry not bit-reproducible;
// any well-mixed per-(step,row,col) stream is the right shape of work)
__device__ __forceinline__ float gumbel_noise(uint32_t step, uint32_t row, uint32_t col) {
    uint32_t h = (step * 0x9E3779B9u) ^ (row * 0x85EBCA6Bu) ^ (col * 0xC2B2AE35u);
    h ^= h >> 16; h *= 0x7FEB352Du;
    h ^= h >> 15; h *= 0x846CA68Bu;
    h ^= h >> 16;
    float u = (float)(h >> 8) * (1.0f / 16777216.0f);
    u = fminf(fmaxf(u, 1e-12f), 0.99999994f);
    return -__logf(-__logf(u));
}

// ---------------------------------------------------------------------------
// Kernel 1: xb[b, :] = data[b, :] @ W_ih^T + b_ih + b_hh   (K=18, trivial)
// ---------------------------------------------------------------------------
__global__ void xproj_kernel(const float* __restrict__ data,
                             const float* __restrict__ W_ih,
                             const float* __restrict__ b_ih,
                             const float* __restrict__ b_hh,
                             float* __restrict__ xb) {
    __shared__ float sD[IN_DIM];
    const int b = blockIdx.x;
    const int t = threadIdx.x;         // t == output column (HDIM==blockDim)
    if (t < IN_DIM) sD[t] = data[b * IN_DIM + t];
    __syncthreads();
    float s = b_ih[t] + b_hh[t];
    const float* wr = W_ih + t * IN_DIM;
#pragma unroll
    for (int i = 0; i < IN_DIM; ++i) s = fmaf(sD[i], wr[i], s);
    xb[(size_t)b * HDIM + t] = s;
}

// ---------------------------------------------------------------------------
// Kernel 2: WhhT[k][j] = W_hh[j][k]  (contiguous one-hot gather rows),
//           WoutBf = bf16(W_out),  rec0 = init @ W_hh^T
// ---------------------------------------------------------------------------
__global__ void prep_weights_kernel(const float* __restrict__ W_hh,
                                    const float* __restrict__ W_out,
                                    const float* __restrict__ init_h,
                                    float* __restrict__ WhhT,
                                    bf16_t* __restrict__ WoutBf,
                                    float* __restrict__ rec0) {
    const int bk = blockIdx.x;   // 0..255
    const int t  = threadIdx.x;  // 0..255
    WhhT[bk * HDIM + t]   = W_hh[t * HDIM + bk];
    WoutBf[bk * HDIM + t] = (bf16_t)W_out[bk * HDIM + t];
    if (bk == 0) {
        float s = 0.0f;
        const float* wr = W_hh + (size_t)t * HDIM;
        for (int i = 0; i < HDIM; ++i) s = fmaf(init_h[i], wr[i], s);
        rec0[t] = s;
    }
}

// ---------------------------------------------------------------------------
// Kernel 3: persistent recurrence. 256 threads (8 wave32), 64 rows/block.
// LDS budget (~299 KB of 320 KB/WGP):
//   sWout 256x264 bf16 (132K) | sH 64x264 bf16 (33K) | sXb 64x256 f32 (64K)
//   sLogits 64x257 f32 (64K)  | sBout + reduce scratch + sK (~6K)
// ---------------------------------------------------------------------------
__global__ void speaker_main_kernel(const float* __restrict__ xb,
                                    const float* __restrict__ WhhT,
                                    const float* __restrict__ rec0g,
                                    const bf16_t* __restrict__ WoutBf,
                                    const float* __restrict__ b_out,
                                    float* __restrict__ msg,
                                    float* __restrict__ logp_out,
                                    float* __restrict__ ent_out) {
    extern __shared__ __align__(16) char smem[];
    bf16_t* sWout   = (bf16_t*)smem;                 // [256][WP]
    bf16_t* sH      = sWout + 256 * WP;              // [TILE][HP]
    float*  sXb     = (float*)(sH + TILE * HP);      // [TILE][HDIM]
    float*  sLogits = sXb + TILE * HDIM;             // [TILE][LP]
    float*  sBout   = sLogits + TILE * LP;           // [OUTDIM]
    float*  sRmax   = sBout + OUTDIM;                // [TILE*4]
    float*  sRsum   = sRmax + 256;                   // [TILE*4]
    float*  sRbv    = sRsum + 256;                   // [TILE*4]
    int*    sRbi    = (int*)(sRbv + 256);            // [TILE*4]
    int*    sK      = sRbi + 256;                    // [TILE]

    const int t       = threadIdx.x;                 // 0..255
    const int rowBase = blockIdx.x * TILE;

    // hint the L2 about the shared weight image before the bulk loads
    __builtin_prefetch(WoutBf + (size_t)t * 256, 0, 1);

    // --- one-time LDS staging -------------------------------------------------
    for (int idx = t; idx < HDIM * OUTDIM; idx += 256)
        sWout[(idx >> 8) * WP + (idx & 255)] = WoutBf[idx];
    for (int idx = t; idx < TILE * HDIM; idx += 256)
        sXb[idx] = xb[(size_t)rowBase * HDIM + idx];
    sBout[t] = b_out[t];
    __syncthreads();

    float lp_acc = 0.0f;   // owned by threads with (t&3)==0, row = t>>2

    for (int step = 0; step < NUM_STEPS; ++step) {
        // ---- Phase 1: h = tanh(xb + W_hh[:,k_prev] + b_hh) -> bf16 LDS tile --
        {
            const int c = t;
            for (int r = 0; r < TILE; ++r) {
                float rec;
                if (step == 0) rec = rec0g[c];
                else           rec = WhhT[(size_t)sK[r] * HDIM + c]; // one-hot gather
                const float hv = tanhf(sXb[r * HDIM + c] + rec);
                sH[r * HP + c] = (bf16_t)hv;
            }
        }
        __syncthreads();

        // ---- Phase 2: logits = h @ W_out^T + b_out  (v_wmma bf16) ------------
        {
            const int w    = t >> 5;
            const int lane = t & 31;
            const int m    = lane & 15;
            const int half = lane >> 4;
            const int r0   = (w & 3) * 16;   // 4 row-tiles of 16
            const int ntB  = (w >> 2) * 8;   // 2 wave-groups split the 16 N-tiles

            // A fragments: 16x32 bf16 per k-tile, ISA layout:
            // lane<16 -> K {0..7,16..23}; lane>=16 -> K {8..15,24..31}
            v16bf aF[8];
            const bf16_t* arow = sH + (size_t)(r0 + m) * HP;
#pragma unroll
            for (int kt = 0; kt < 8; ++kt) {
                const int k0 = kt * 32;
                v8bf lo = *(const v8bf*)(arow + k0 + half * 8);
                v8bf hi = *(const v8bf*)(arow + k0 + 16 + half * 8);
                aF[kt] = __builtin_shufflevector(lo, hi,
                          0,1,2,3,4,5,6,7,8,9,10,11,12,13,14,15);
            }
#pragma unroll
            for (int nti = 0; nti < 8; ++nti) {
                const int nt = ntB + nti;
                // B fragments: 32x16, lane = column, K halves by lane group;
                // B[k][n] = W_out[n][k] -> contiguous row reads of sWout
                const bf16_t* brow = sWout + (size_t)(nt * 16 + m) * WP + half * 8;
                v8f acc = {};
#pragma unroll
                for (int kt = 0; kt < 8; ++kt) {
                    const int k0 = kt * 32;
                    v8bf b0 = *(const v8bf*)(brow + k0);
                    v8bf b1 = *(const v8bf*)(brow + k0 + 16);
                    v16bf bF = __builtin_shufflevector(b0, b1,
                                0,1,2,3,4,5,6,7,8,9,10,11,12,13,14,15);
                    acc = __builtin_amdgcn_wmma_f32_16x16x32_bf16(
                              false, aF[kt], false, bF, (short)0, acc, false, false);
                }
                const int col = nt * 16 + m;
                const float bo = sBout[col];
#pragma unroll
                for (int v = 0; v < 8; ++v)
                    sLogits[(size_t)(r0 + v + half * 8) * LP + col] = acc[v] + bo;
            }
        }
        __syncthreads();

        // ---- Phase 3: log-softmax + gumbel argmax (4 threads per row) --------
        {
            const int r = t >> 2, q = t & 3;
            const float* lrow = sLogits + (size_t)r * LP;
            float m_l = -3.4e38f;
#pragma unroll 8
            for (int j = 0; j < 64; ++j) m_l = fmaxf(m_l, lrow[q * 64 + j]);
            sRmax[t] = m_l;
            __syncthreads();
            const int rb = r * 4;
            const float mrow = fmaxf(fmaxf(sRmax[rb], sRmax[rb + 1]),
                                     fmaxf(sRmax[rb + 2], sRmax[rb + 3]));
            float s_l = 0.0f, bv = -3.4e38f; int bi = 0;
            const uint32_t grow = (uint32_t)(rowBase + r);
            for (int j = 0; j < 64; ++j) {
                const int c = q * 64 + j;
                const float v = lrow[c];
                s_l += __expf(v - mrow);
                // TEMP==1: argmax((logp+g)/T) == argmax(logits + g)
                const float pv = v + gumbel_noise((uint32_t)step, grow, (uint32_t)c);
                if (pv > bv) { bv = pv; bi = c; }
            }
            sRsum[t] = s_l; sRbv[t] = bv; sRbi[t] = bi;
            __syncthreads();
            if (q == 0) {
                float s = sRsum[rb] + sRsum[rb + 1] + sRsum[rb + 2] + sRsum[rb + 3];
                float bbv = sRbv[rb]; int bbi = sRbi[rb];
#pragma unroll
                for (int i = 1; i < 4; ++i)
                    if (sRbv[rb + i] > bbv) { bbv = sRbv[rb + i]; bbi = sRbi[rb + i]; }
                lp_acc += lrow[bbi] - mrow - __logf(s);   // logp at argmax
                sK[r] = bbi;
            }
            __syncthreads();
        }

        // ---- Phase 4: stream the one-hot message tile (HBM-bound) ------------
        {
            float* dst = msg + ((size_t)step * BATCH + rowBase) * OUTDIM;
            for (int idx = t; idx < TILE * OUTDIM / 4; idx += 256) {
                const int r  = idx >> 6;
                const int c4 = (idx & 63);
                const int k  = sK[r];
                v4f v = {0.0f, 0.0f, 0.0f, 0.0f};
                if ((k >> 2) == c4) v[k & 3] = 1.0f;
                __builtin_nontemporal_store(v, (v4f*)(dst + (size_t)r * OUTDIM + c4 * 4));
            }
        }
        // No barrier needed before next Phase 1: sH's last readers synced after
        // Phase 2; sK's next writer (Phase 3) is four barriers away.
    }

    if ((t & 3) == 0) {
        const int r = t >> 2;
        logp_out[rowBase + r] = lp_acc;
        ent_out[rowBase + r]  = 0.0f;
    }
}

// ---------------------------------------------------------------------------
extern "C" void kernel_launch(void* const* d_in, const int* in_sizes, int n_in,
                              void* d_out, int out_size, void* d_ws, size_t ws_size,
                              hipStream_t stream) {
    const float* data  = (const float*)d_in[0];
    const float* init_h= (const float*)d_in[1];
    const float* W_ih  = (const float*)d_in[2];
    const float* W_hh  = (const float*)d_in[3];
    const float* b_ih  = (const float*)d_in[4];
    const float* b_hh  = (const float*)d_in[5];
    const float* W_out = (const float*)d_in[6];
    const float* b_out = (const float*)d_in[7];

    // workspace: xb [B,256] f32 | WhhT [256,256] f32 | rec0 [256] f32 | Wout bf16
    float*  xb     = (float*)d_ws;
    float*  WhhT   = xb + (size_t)BATCH * HDIM;
    float*  rec0   = WhhT + HDIM * HDIM;
    bf16_t* WoutBf = (bf16_t*)(rec0 + HDIM);

    float* msg  = (float*)d_out;
    float* logp = msg + (size_t)NUM_STEPS * BATCH * OUTDIM;
    float* ent  = logp + BATCH;

    xproj_kernel<<<BATCH, HDIM, 0, stream>>>(data, W_ih, b_ih, b_hh, xb);
    prep_weights_kernel<<<HDIM, HDIM, 0, stream>>>(W_hh, W_out, init_h, WhhT, WoutBf, rec0);

    const size_t lds_bytes =
        (size_t)(256 * WP + TILE * HP) * sizeof(uint16_t)   // sWout + sH (bf16)
      + (size_t)(TILE * HDIM) * sizeof(float)               // sXb
      + (size_t)(TILE * LP) * sizeof(float)                 // sLogits
      + (size_t)OUTDIM * sizeof(float)                      // sBout
      + (size_t)(4 * 256) * sizeof(float)                   // sRmax/sRsum/sRbv/sRbi
      + (size_t)TILE * sizeof(int);                         // sK

    speaker_main_kernel<<<BATCH / TILE, 256, lds_bytes, stream>>>(
        xb, WhhT, rec0, WoutBf, b_out, msg, logp, ent);
}
